// HierarchicalSynthesisNet_22239340658913
// MI455X (gfx1250) — matrix operand order repaired and checked by reading, not compiled, not run on hardware
//
#include <hip/hip_runtime.h>
#include <hip/hip_bf16.h>
#include <math.h>

typedef __attribute__((ext_vector_type(16))) __bf16 bf16x16;
typedef __attribute__((ext_vector_type(8)))  float  floatx8;

#define DEV __device__ __forceinline__

DEV unsigned short f2bf(float f) {
  unsigned u = __float_as_uint(f);
  u += 0x7FFFu + ((u >> 16) & 1u);
  return (unsigned short)(u >> 16);
}
DEV float bf2f(unsigned short h) { return __uint_as_float(((unsigned)h) << 16); }
DEV float lrelu(float v) { return v > 0.f ? v : 0.2f * v; }
DEV unsigned pcg(unsigned x) {
  x = x * 747796405u + 2891336453u;
  x = ((x >> ((x >> 28) + 4u)) ^ x) * 277803737u;
  return (x >> 22) ^ x;
}
DEV float gauss(unsigned seed) {
  unsigned a = pcg(seed);
  unsigned b = pcg(seed ^ 0x9E3779B9u);
  float u1 = ((float)a + 1.0f) * 2.3283064365386963e-10f;  // (0,1]
  float u2 = (float)b * 2.3283064365386963e-10f;
  return __fsqrt_rn(fmaxf(-2.f * __logf(u1), 0.f)) * __cosf(6.28318530718f * u2);
}

// ---------------- small VALU kernels ----------------

__global__ void k_mean64(const float* __restrict__ z, float* __restrict__ out, int n) {
  int idx = blockIdx.x * blockDim.x + threadIdx.x;
  if (idx >= n) return;
  float acc = 0.f;
  const float* p = z + (size_t)idx * 64;
  for (int j = 0; j < 64; ++j) acc += p[j];
  out[idx] = acc * (1.f / 64.f);
}

__global__ void k_pool2(const float* __restrict__ in, float* __restrict__ out,
                        int B, int C, int H, int W) {
  int n = B * C * H * W;
  int idx = blockIdx.x * blockDim.x + threadIdx.x;
  if (idx >= n) return;
  int x = idx % W; int t = idx / W;
  int y = t % H;   t /= H;
  const float* base = in + (size_t)t * (4 * H * W) + (size_t)(2 * y) * (2 * W) + 2 * x;
  out[idx] = 0.25f * (base[0] + base[1] + base[2 * W] + base[2 * W + 1]);
}

__global__ void k_fc(const float* __restrict__ in, const float* __restrict__ w,
                     const float* __restrict__ bias, float* __restrict__ out,
                     int B, int IN, int OUT, int act) {
  int idx = blockIdx.x * blockDim.x + threadIdx.x;
  if (idx >= B * OUT) return;
  int o = idx % OUT, b = idx / OUT;
  float acc = bias ? bias[o] : 0.f;
  const float* ib = in + (size_t)b * IN;
  const float* wo = w + (size_t)o * IN;
  for (int i = 0; i < IN; ++i) acc = fmaf(ib[i], wo[i], acc);
  out[idx] = act ? lrelu(acc) : acc;
}

__global__ void k_prep_conv(const float* __restrict__ W, unsigned short* __restrict__ wbf,
                            float* __restrict__ w2, int O, int I) {
  int idx = blockIdx.x * blockDim.x + threadIdx.x;  // o*I + i
  if (idx >= O * I) return;
  float sc = rsqrtf((float)(I * 9));
  float acc = 0.f;
  const float* src = W + (size_t)idx * 9;
  unsigned short* dst = wbf + (size_t)idx * 9;
  for (int r = 0; r < 9; ++r) {
    float v = src[r] * sc;
    dst[r] = f2bf(v);
    acc = fmaf(v, v, acc);
  }
  w2[idx] = acc;
}

__global__ void k_prep_rgb(const float* __restrict__ W, float* __restrict__ ws, int I) {
  int idx = blockIdx.x * blockDim.x + threadIdx.x;
  if (idx >= 3 * I) return;
  ws[idx] = W[idx] * rsqrtf((float)I);
}

__global__ void k_demod(const float* __restrict__ w2, const float* __restrict__ s,
                        float* __restrict__ d, int B, int O, int I) {
  int idx = blockIdx.x * blockDim.x + threadIdx.x;
  if (idx >= B * O) return;
  int o = idx % O, b = idx / O;
  float acc = 0.f;
  const float* wo = w2 + (size_t)o * I;
  const float* sb = s + (size_t)b * I;
  for (int i = 0; i < I; ++i) acc = fmaf(wo[i], sb[i] * sb[i], acc);
  d[idx] = rsqrtf(acc + 1e-8f);
}

__global__ void k_const_init(const float* __restrict__ cst, unsigned short* __restrict__ x, int B) {
  int n = B * 512 * 16;
  int idx = blockIdx.x * blockDim.x + threadIdx.x;
  if (idx >= n) return;
  x[idx] = f2bf(cst[idx % (512 * 16)]);
}

__global__ void k_inject_add(unsigned short* __restrict__ x, const float* __restrict__ z,
                             const float* __restrict__ w, const float* __restrict__ bias,
                             int B, int C, int HW) {
  int n = B * C * HW;
  int idx = blockIdx.x * blockDim.x + threadIdx.x;
  if (idx >= n) return;
  int p = idx % HW; int bc = idx / HW; int o = bc % C; int b = bc / C;
  float acc = bias[o];
  const float* wo = w + (size_t)o * 64;
  const float* zb = z + (size_t)b * 64 * HW + p;
  for (int i = 0; i < 64; ++i) acc = fmaf(wo[i], zb[(size_t)i * HW], acc);
  x[idx] = f2bf(bf2f(x[idx]) + acc);
}

__global__ void k_up_bf(const unsigned short* __restrict__ in, unsigned short* __restrict__ out,
                        int B, int C, int H, int W) {
  int OH = 2 * H, OW = 2 * W;
  int n = B * C * OH * OW;
  int idx = blockIdx.x * blockDim.x + threadIdx.x;
  if (idx >= n) return;
  int ox = idx % OW; int t = idx / OW;
  int oy = t % OH;   t /= OH;
  float sy = 0.5f * oy - 0.25f, sx = 0.5f * ox - 0.25f;
  int y0 = (int)floorf(sy), x0 = (int)floorf(sx);
  float fy = sy - y0, fx = sx - x0;
  int y0c = min(max(y0, 0), H - 1), y1c = min(max(y0 + 1, 0), H - 1);
  int x0c = min(max(x0, 0), W - 1), x1c = min(max(x0 + 1, 0), W - 1);
  const unsigned short* base = in + (size_t)t * H * W;
  float v00 = bf2f(base[y0c * W + x0c]), v01 = bf2f(base[y0c * W + x1c]);
  float v10 = bf2f(base[y1c * W + x0c]), v11 = bf2f(base[y1c * W + x1c]);
  float v = (1.f - fy) * ((1.f - fx) * v00 + fx * v01) + fy * ((1.f - fx) * v10 + fx * v11);
  out[idx] = f2bf(v);
}

__global__ void k_up_f32(const float* __restrict__ in, float* __restrict__ out,
                         int B, int C, int H, int W) {
  int OH = 2 * H, OW = 2 * W;
  int n = B * C * OH * OW;
  int idx = blockIdx.x * blockDim.x + threadIdx.x;
  if (idx >= n) return;
  int ox = idx % OW; int t = idx / OW;
  int oy = t % OH;   t /= OH;
  float sy = 0.5f * oy - 0.25f, sx = 0.5f * ox - 0.25f;
  int y0 = (int)floorf(sy), x0 = (int)floorf(sx);
  float fy = sy - y0, fx = sx - x0;
  int y0c = min(max(y0, 0), H - 1), y1c = min(max(y0 + 1, 0), H - 1);
  int x0c = min(max(x0, 0), W - 1), x1c = min(max(x0 + 1, 0), W - 1);
  const float* base = in + (size_t)t * H * W;
  float v00 = base[y0c * W + x0c], v01 = base[y0c * W + x1c];
  float v10 = base[y1c * W + x0c], v11 = base[y1c * W + x1c];
  out[idx] = (1.f - fy) * ((1.f - fx) * v00 + fx * v01) + fy * ((1.f - fx) * v10 + fx * v11);
}

__global__ void k_torgb(const unsigned short* __restrict__ x, const float* __restrict__ w,
                        const float* __restrict__ s, const float* __restrict__ nsc,
                        float* __restrict__ rgb, int accumulate,
                        int B, int I, int HW, unsigned seed) {
  int idx = blockIdx.x * blockDim.x + threadIdx.x;  // b*HW + p
  if (idx >= B * HW) return;
  int b = idx / HW, p = idx - b * HW;
  float a0 = 0.f, a1 = 0.f, a2 = 0.f;
  const unsigned short* xb = x + (size_t)b * I * HW + p;
  const float* sb = s + (size_t)b * I;
  for (int i = 0; i < I; ++i) {
    float xs = bf2f(xb[(size_t)i * HW]) * sb[i];
    a0 = fmaf(w[i], xs, a0);
    a1 = fmaf(w[I + i], xs, a1);
    a2 = fmaf(w[2 * I + i], xs, a2);
  }
  float nz = gauss(seed ^ (unsigned)idx) * nsc[0];
  float v0 = lrelu(a0 + nz), v1 = lrelu(a1 + nz), v2 = lrelu(a2 + nz);
  size_t o0 = ((size_t)b * 3 + 0) * HW + p;
  size_t o1 = ((size_t)b * 3 + 1) * HW + p;
  size_t o2 = ((size_t)b * 3 + 2) * HW + p;
  if (accumulate) { rgb[o0] += v0; rgb[o1] += v1; rgb[o2] += v2; }
  else            { rgb[o0]  = v0; rgb[o1]  = v1; rgb[o2]  = v2; }
}

__global__ void k_tanh(const float* __restrict__ in, float* __restrict__ out, int n) {
  int idx = blockIdx.x * blockDim.x + threadIdx.x;
  if (idx < n) out[idx] = tanhf(in[idx]);
}

// ---------------- WMMA implicit-GEMM modulated 3x3 conv ----------------
// Block: 128 threads (4 waves, 2x2), tile 64(M=o) x 64(N=pix), K step 64
// (two K=32 halves). Each wave: 2x2 fragment tiles -> 4 f32 accumulators,
// 8 WMMAs per K step with A/B fragment reuse. Panels are staged in
// *fragment order* [kc][tile][lane][elem] so that:
//   - A staging: one global b128 load (8 contiguous bf16 weights) + one LDS
//     b128 store (the swizzle maps 8 consecutive k to 8 consecutive elems)
//   - B staging: 8 im2col gathers + one LDS b128 store
//   - fragment loads: two contiguous ds_load_b128 per lane
__global__ void k_modconv_wmma(const unsigned short* __restrict__ xin,
                               const unsigned short* __restrict__ wbf,
                               const float* __restrict__ s,
                               const float* __restrict__ dmod,
                               const float* __restrict__ nsc,
                               unsigned short* __restrict__ xout,
                               int Cin, int Cout, int H, int W,
                               int shW, int mskW, unsigned seed) {
  const int HW = H * W;
  const int K = Cin * 9;                 // divisible by 64 for all layers
  const int b = blockIdx.z;
  const int oBase = blockIdx.y * 64;
  const int nBase = blockIdx.x * 64;
  const int tid = threadIdx.x;
  const int lane = tid & 31;
  const int wave = tid >> 5;
  const int wm = wave >> 1, wn = wave & 1;
  const int hi = lane >> 4;

  __shared__ __align__(16) unsigned short sAf[2][4][32][16];  // 8 KB
  __shared__ __align__(16) unsigned short sBf[2][4][32][16];  // 8 KB

  floatx8 acc[2][2];
#pragma unroll
  for (int i = 0; i < 2; ++i)
#pragma unroll
    for (int j = 0; j < 2; ++j)
#pragma unroll
      for (int r = 0; r < 8; ++r) acc[i][j][r] = 0.f;

  const float* sb = s + (size_t)b * Cin;
  const unsigned short* xb = xin + (size_t)b * Cin * HW;

  union U16 { bf16x16 v; uint4 q[2]; };
  union U8  { uint4 q; unsigned short u[8]; };

  for (int kk = 0; kk < K; kk += 64) {
    if (kk + 64 < K)
      __builtin_prefetch(&wbf[(size_t)(oBase + (tid & 63)) * K + kk + 64], 0, 1);

    // stage A: 64 m rows x 64 k, vectorized 8-k chunks
    for (int c = tid; c < 64 * 8; c += 128) {
      int m = c >> 3, kg = c & 7;        // k = kg*8 + j
      int o = oBase + m;
      U8 w8;
      if (o < Cout) w8.q = *(const uint4*)&wbf[(size_t)o * K + kk + (kg << 3)];
      else { w8.q.x = 0; w8.q.y = 0; w8.q.z = 0; w8.q.w = 0; }
      // A-swizzle: e = ((kg>>1)&1)*8 + j (contiguous), laneHi = kg&1
      int kc = kg >> 2;
      int ebase = ((kg >> 1) & 1) << 3;
      int lnd = (m & 15) | ((kg & 1) << 4);
      *(uint4*)&sAf[kc][m >> 4][lnd][ebase] = w8.q;
    }
    // stage B: im2col of modulated input, 64 k x 64 n, 8-k chunks per store
    for (int c = tid; c < 64 * 8; c += 128) {
      int nn = c >> 3, kg = c & 7;
      int gn = nBase + nn;
      U8 v8;
#pragma unroll
      for (int j = 0; j < 8; ++j) {
        int gk = kk + (kg << 3) + j;
        float v = 0.f;
        if (gn < HW) {
          int ci = gk / 9, r = gk - ci * 9;
          int ky = r / 3, kx = r - ky * 3;
          int y = (gn >> shW) + ky - 1;
          int x = (gn & mskW) + kx - 1;
          if ((unsigned)y < (unsigned)H && (unsigned)x < (unsigned)W)
            v = bf2f(xb[(size_t)ci * HW + (y << shW) + x]) * sb[ci];
        }
        v8.u[j] = f2bf(v);
      }
      // B-swizzle: e = ((kg>>1)&1)*8 + j (contiguous), laneHi = kg&1
      int kc = kg >> 2;
      int ebase = ((kg >> 1) & 1) << 3;
      int lnd = (nn & 15) | ((kg & 1) << 4);
      *(uint4*)&sBf[kc][nn >> 4][lnd][ebase] = v8.q;
    }
    __syncthreads();

#pragma unroll
    for (int kc = 0; kc < 2; ++kc) {
      U16 a0, a1, b0, b1;
      const uint4* pa0 = (const uint4*)&sAf[kc][(wm << 1) | 0][lane][0];
      const uint4* pa1 = (const uint4*)&sAf[kc][(wm << 1) | 1][lane][0];
      const uint4* pb0 = (const uint4*)&sBf[kc][(wn << 1) | 0][lane][0];
      const uint4* pb1 = (const uint4*)&sBf[kc][(wn << 1) | 1][lane][0];
      a0.q[0] = pa0[0]; a0.q[1] = pa0[1];
      a1.q[0] = pa1[0]; a1.q[1] = pa1[1];
      b0.q[0] = pb0[0]; b0.q[1] = pb0[1];
      b1.q[0] = pb1[0]; b1.q[1] = pb1[1];
      acc[0][0] = __builtin_amdgcn_wmma_f32_16x16x32_bf16(false, a0.v, false, b0.v, (short)0, acc[0][0], false, false);
      acc[0][1] = __builtin_amdgcn_wmma_f32_16x16x32_bf16(false, a0.v, false, b1.v, (short)0, acc[0][1], false, false);
      acc[1][0] = __builtin_amdgcn_wmma_f32_16x16x32_bf16(false, a1.v, false, b0.v, (short)0, acc[1][0], false, false);
      acc[1][1] = __builtin_amdgcn_wmma_f32_16x16x32_bf16(false, a1.v, false, b1.v, (short)0, acc[1][1], false, false);
    }
    __syncthreads();
  }

  // epilogue: demod, noise (shared across channels), leaky, bf16 store
  float ns = nsc[0];
#pragma unroll
  for (int nt = 0; nt < 2; ++nt) {
    int n = nBase + (wn << 5) + (nt << 4) + (lane & 15);
    if (n >= HW) continue;
    float nz = gauss(seed ^ (unsigned)(b * HW + n)) * ns;
#pragma unroll
    for (int mt = 0; mt < 2; ++mt) {
#pragma unroll
      for (int r = 0; r < 8; ++r) {
        int o = oBase + (wm << 5) + (mt << 4) + r + (hi << 3);
        if (o < Cout) {
          float v = acc[mt][nt][r];
          if (dmod) v *= dmod[(size_t)b * Cout + o];
          v = lrelu(v + nz);
          xout[((size_t)b * Cout + o) * HW + n] = f2bf(v);
        }
      }
    }
  }
}

// ---------------- host orchestration ----------------
static inline size_t alignup(size_t x) { return (x + 255) & ~(size_t)255; }
static inline int ilog2(int v) { int s = 0; while ((1 << s) < v) ++s; return s; }

extern "C" void kernel_launch(void* const* d_in, const int* in_sizes, int n_in,
                              void* d_out, int out_size, void* d_ws, size_t ws_size,
                              hipStream_t stream) {
  (void)in_sizes; (void)n_in; (void)out_size; (void)ws_size;
  const int B = 32;
  const float* z_t = (const float*)d_in[0];
  const float* z_m = (const float*)d_in[1];
  const float* z_b = (const float*)d_in[2];
  const float* map_w1 = (const float*)d_in[3];
  const float* map_b1 = (const float*)d_in[4];
  const float* map_w2 = (const float*)d_in[5];
  const float* map_b2 = (const float*)d_in[6];
  const float* cst    = (const float*)d_in[7];
  const float* inj_mid_w = (const float*)d_in[56];
  const float* inj_mid_b = (const float*)d_in[57];
  const float* inj_bot_w = (const float*)d_in[58];
  const float* inj_bot_b = (const float*)d_in[59];

  const int cI[6] = {512, 512, 512, 512, 256, 128};
  const int cO[6] = {512, 512, 512, 256, 128, 64};
  const int rI[6] = {512, 512, 512, 256, 128, 64};
  const int Hs[6] = {4, 8, 16, 32, 64, 128};

  char* base = (char*)d_ws;
  size_t off = 0;
  auto alloc = [&](size_t bytes) -> void* { void* p = base + off; off = alignup(off + bytes); return p; };

  unsigned short* xA = (unsigned short*)alloc((size_t)B * 128 * 128 * 128 * sizeof(unsigned short));
  unsigned short* xB = (unsigned short*)alloc((size_t)B * 128 * 128 * 128 * sizeof(unsigned short));
  float* rgbA = (float*)alloc((size_t)B * 3 * 128 * 128 * sizeof(float));
  float* rgbB = (float*)alloc((size_t)B * 3 * 128 * 128 * sizeof(float));

  unsigned short* wbf[6]; float* w2[6]; float* wrgb[6];
  float* sConv[6]; float* sRgb[6]; float* dMod[6];
  for (int l = 0; l < 6; ++l) {
    wbf[l] = (unsigned short*)alloc((size_t)cO[l] * cI[l] * 9 * sizeof(unsigned short));
    w2[l]  = (float*)alloc((size_t)cO[l] * cI[l] * sizeof(float));
    wrgb[l] = (float*)alloc((size_t)3 * rI[l] * sizeof(float));
    sConv[l] = (float*)alloc((size_t)B * cI[l] * sizeof(float));
    sRgb[l]  = (float*)alloc((size_t)B * rI[l] * sizeof(float));
    dMod[l]  = (float*)alloc((size_t)B * cO[l] * sizeof(float));
  }
  float* zp   = (float*)alloc((size_t)B * 64 * sizeof(float));
  float* hbuf = (float*)alloc((size_t)B * 512 * sizeof(float));
  float* wlat = (float*)alloc((size_t)B * 512 * sizeof(float));
  float* zm8  = (float*)alloc((size_t)B * 64 * 8 * 8 * sizeof(float));

  auto g1 = [](int n) { return dim3((unsigned)((n + 255) / 256)); };

  for (int l = 0; l < 6; ++l) {
    int n = cO[l] * cI[l];
    k_prep_conv<<<g1(n), 256, 0, stream>>>((const float*)d_in[8 + 4 * l], wbf[l], w2[l], cO[l], cI[l]);
    k_prep_rgb<<<g1(3 * rI[l]), 256, 0, stream>>>((const float*)d_in[32 + 4 * l], wrgb[l], rI[l]);
  }

  k_mean64<<<g1(B * 64), 256, 0, stream>>>(z_t, zp, B * 64);
  k_fc<<<g1(B * 512), 256, 0, stream>>>(zp, map_w1, map_b1, hbuf, B, 64, 512, 1);
  k_fc<<<g1(B * 512), 256, 0, stream>>>(hbuf, map_w2, map_b2, wlat, B, 512, 512, 1);

  for (int l = 0; l < 6; ++l) {
    k_fc<<<g1(B * cI[l]), 256, 0, stream>>>(wlat, (const float*)d_in[9 + 4 * l],
                                            (const float*)d_in[10 + 4 * l], sConv[l], B, 512, cI[l], 0);
    k_fc<<<g1(B * rI[l]), 256, 0, stream>>>(wlat, (const float*)d_in[33 + 4 * l],
                                            (const float*)d_in[34 + 4 * l], sRgb[l], B, 512, rI[l], 0);
    k_demod<<<g1(B * cO[l]), 256, 0, stream>>>(w2[l], sConv[l], dMod[l], B, cO[l], cI[l]);
  }

  k_pool2<<<g1(B * 64 * 8 * 8), 256, 0, stream>>>(z_m, zm8, B, 64, 8, 8);

  auto conv = [&](int l, const unsigned short* in, unsigned short* out, unsigned seed) {
    int H = Hs[l], W = Hs[l];
    dim3 grid((unsigned)((H * W + 63) / 64), (unsigned)((cO[l] + 63) / 64), (unsigned)B);
    k_modconv_wmma<<<grid, 128, 0, stream>>>(in, wbf[l], sConv[l], dMod[l],
                                             (const float*)d_in[11 + 4 * l], out,
                                             cI[l], cO[l], H, W, ilog2(W), W - 1,
                                             0xABCD0000u + seed);
  };
  auto torgb = [&](int l, const unsigned short* in, float* rgb, int accum, unsigned seed) {
    int HW = Hs[l] * Hs[l];
    k_torgb<<<g1(B * HW), 256, 0, stream>>>(in, wrgb[l], sRgb[l],
                                            (const float*)d_in[35 + 4 * l], rgb, accum,
                                            B, rI[l], HW, 0xABCD0000u + seed);
  };

  k_const_init<<<g1(B * 512 * 16), 256, 0, stream>>>(cst, xA, B);
  conv(0, xA, xB, 0);
  torgb(0, xB, rgbA, 0, 1);
  k_up_bf<<<g1(B * 512 * 64), 256, 0, stream>>>(xB, xA, B, 512, 4, 4);
  k_up_f32<<<g1(B * 3 * 64), 256, 0, stream>>>(rgbA, rgbB, B, 3, 4, 4);

  k_inject_add<<<g1(B * 512 * 64), 256, 0, stream>>>(xA, zm8, inj_mid_w, inj_mid_b, B, 512, 64);
  conv(1, xA, xB, 2);
  torgb(1, xB, rgbB, 1, 3);
  k_up_bf<<<g1(B * 512 * 256), 256, 0, stream>>>(xB, xA, B, 512, 8, 8);
  k_up_f32<<<g1(B * 3 * 256), 256, 0, stream>>>(rgbB, rgbA, B, 3, 8, 8);

  conv(2, xA, xB, 4);
  torgb(2, xB, rgbA, 1, 5);
  k_up_bf<<<g1(B * 512 * 1024), 256, 0, stream>>>(xB, xA, B, 512, 16, 16);
  k_up_f32<<<g1(B * 3 * 1024), 256, 0, stream>>>(rgbA, rgbB, B, 3, 16, 16);

  k_inject_add<<<g1(B * 512 * 1024), 256, 0, stream>>>(xA, z_b, inj_bot_w, inj_bot_b, B, 512, 1024);
  conv(3, xA, xB, 6);
  torgb(3, xB, rgbB, 1, 7);
  k_up_bf<<<g1(B * 256 * 4096), 256, 0, stream>>>(xB, xA, B, 256, 32, 32);
  k_up_f32<<<g1(B * 3 * 4096), 256, 0, stream>>>(rgbB, rgbA, B, 3, 32, 32);

  conv(4, xA, xB, 8);
  torgb(4, xB, rgbA, 1, 9);
  k_up_bf<<<g1(B * 128 * 16384), 256, 0, stream>>>(xB, xA, B, 128, 64, 64);
  k_up_f32<<<g1(B * 3 * 16384), 256, 0, stream>>>(rgbA, rgbB, B, 3, 64, 64);

  conv(5, xA, xB, 10);
  torgb(5, xB, rgbB, 1, 11);

  k_tanh<<<g1(B * 3 * 16384), 256, 0, stream>>>(rgbB, (float*)d_out, B * 3 * 16384);
}
